// MonarchLinear_1185410974364
// MI455X (gfx1250) — matrix operand order repaired and checked by reading, not compiled, not run on hardware
//
#include <hip/hip_runtime.h>

// ---------------------------------------------------------------------------
// Monarch linear for MI455X (gfx1250, wave32, WMMA).
// Two GEMM stages via split-bf16 (hi+lo) emulation: 3x v_wmma_f32_16x16x32_bf16
// per 16x16x32 step gives ~fp32 accuracy at ~2.7x the fp32-WMMA rate.
// Round 2: double-buffered LDS + register-staged global prefetch, one barrier
// per K-chunk so global-load latency hides behind 24 WMMAs.
// ---------------------------------------------------------------------------

typedef __bf16 bf16_t;
typedef __attribute__((ext_vector_type(8)))  bf16_t       v8bf;
typedef __attribute__((ext_vector_type(16))) bf16_t       v16bf;
typedef __attribute__((ext_vector_type(8)))  float        v8f;
typedef __attribute__((ext_vector_type(4)))  float        v4f;
typedef __attribute__((ext_vector_type(4)))  unsigned int v4u;

#define N_TOT   4096
#define BLK_K   1024
#define NBLK    4
#define BM      128
#define BN      128
#define KC      32
#define NCHUNK  (BLK_K / KC)
#define AS      40   // LDS row stride (32 data + 8 pad) bf16 -> 80B, 16B aligned

union Frag { v16bf v; v8bf h[2]; };

static __device__ inline v8f wmma_bf16(v16bf a, v16bf b, v8f c) {
  // (neg_a, A, neg_b, B, c_mod, C, reuse_a, reuse_b)
  return __builtin_amdgcn_wmma_f32_16x16x32_bf16(false, a, false, b, (short)0, c,
                                                 false, false);
}

// A fragment, 16x32 bf16: lanes 0-15 = M rows, K runs {0..7,16..23};
// lanes 16-31 = same M, K runs {8..15,24..31}.
static __device__ inline Frag frag_a(const bf16_t* s, int rowBase, int lane) {
  const int m  = rowBase + (lane & 15);
  const int kh = (lane >> 4) << 3;  // 0 or 8
  Frag f;
  f.h[0] = *(const v8bf*)(s + m * AS + kh);
  f.h[1] = *(const v8bf*)(s + m * AS + 16 + kh);
  return f;
}

// B fragment, 32x16 bf16 (column per lane): lanes 0-15 = N cols, K runs {0..7,16..23};
// lanes 16-31 = same N cols, K runs {8..15,24..31}.
static __device__ inline Frag frag_b(const bf16_t* s, int colBase, int lane) {
  const int n  = colBase + (lane & 15);
  const int kb = (lane >> 4) << 3;  // 0 or 8
  Frag f;
  f.h[0] = *(const v8bf*)(s + n * AS + kb);
  f.h[1] = *(const v8bf*)(s + n * AS + kb + 16);
  return f;
}

static __device__ inline void split_store(bf16_t* hi, bf16_t* lo, int idx, float f) {
  bf16_t h = (bf16_t)f;
  hi[idx] = h;
  lo[idx] = (bf16_t)(f - (float)h);
}

// ---------------------------------------------------------------------------
// Stage 1: out1[b, k*1024+q] = sum_p x[b,k*1024+p] * w1[k,q,p]
// stored permuted+packed: t[b][ (q&3)*1024 + k*256 + (q>>2) ] = {bf16 hi, bf16 lo}
// ---------------------------------------------------------------------------
__global__ __launch_bounds__(256)
void monarch_stage1_bf16x3(const float* __restrict__ x,
                           const float* __restrict__ w1,
                           unsigned int* __restrict__ t)
{
  __shared__ bf16_t sAhi[2][BM * AS], sAlo[2][BM * AS];
  __shared__ bf16_t sBhi[2][BN * AS], sBlo[2][BN * AS];

  const int tid  = threadIdx.x;
  const int lane = tid & 31;
  const int wave = tid >> 5;
  const int wm   = wave >> 2;   // 0..1 : 64-row M slice
  const int wn   = wave & 3;    // 0..3 : 32-col N slice

  const int qt   = blockIdx.x;  // N tile (0..7)
  const int mt   = blockIdx.y;  // M tile
  const int kblk = blockIdx.z;  // monarch block (0..3)

  const float* xg = x  + (size_t)mt * BM * N_TOT + (size_t)kblk * BLK_K;
  const float* wg = w1 + (size_t)kblk * BLK_K * BLK_K + (size_t)qt * BN * BLK_K;

  const int ldRow = tid >> 3;           // 0..31 base row group (i>>3)
  const int ldCol = (tid & 7) << 2;     // 0,4,...,28

  const v8f vzero = {0.f, 0.f, 0.f, 0.f, 0.f, 0.f, 0.f, 0.f};
  v8f acc[4][2];
#pragma unroll
  for (int i = 0; i < 4; ++i)
#pragma unroll
    for (int j = 0; j < 2; ++j) acc[i][j] = vzero;

  v4f aR[4], bR[4];

  auto gload = [&](int kb) {
#pragma unroll
    for (int it = 0; it < 4; ++it) {
      int row = ldRow + it * 32;
      aR[it] = *(const v4f*)(xg + (size_t)row * N_TOT + kb + ldCol);
      bR[it] = *(const v4f*)(wg + (size_t)row * BLK_K + kb + ldCol);
    }
  };
  auto lstore = [&](int buf) {
#pragma unroll
    for (int it = 0; it < 4; ++it) {
      int idx = (ldRow + it * 32) * AS + ldCol;
#pragma unroll
      for (int j = 0; j < 4; ++j) {
        split_store(sAhi[buf], sAlo[buf], idx + j, aR[it][j]);
        split_store(sBhi[buf], sBlo[buf], idx + j, bR[it][j]);
      }
    }
  };
  auto compute = [&](int buf) {
    Frag bhi[2], blo[2];
#pragma unroll
    for (int ns = 0; ns < 2; ++ns) {
      bhi[ns] = frag_b(sBhi[buf], wn * 32 + ns * 16, lane);
      blo[ns] = frag_b(sBlo[buf], wn * 32 + ns * 16, lane);
    }
#pragma unroll
    for (int ms = 0; ms < 4; ++ms) {
      Frag ahi = frag_a(sAhi[buf], wm * 64 + ms * 16, lane);
      Frag alo = frag_a(sAlo[buf], wm * 64 + ms * 16, lane);
#pragma unroll
      for (int ns = 0; ns < 2; ++ns) {
        acc[ms][ns] = wmma_bf16(ahi.v, bhi[ns].v, acc[ms][ns]);
        acc[ms][ns] = wmma_bf16(ahi.v, blo[ns].v, acc[ms][ns]);
        acc[ms][ns] = wmma_bf16(alo.v, bhi[ns].v, acc[ms][ns]);
      }
    }
  };

  gload(0);
  lstore(0);
  __syncthreads();
  for (int c = 0; c < NCHUNK; ++c) {
    const int nxt = c + 1;
    if (nxt < NCHUNK) gload(nxt * KC);    // prefetch next chunk (latency hidden)
    compute(c & 1);
    if (nxt < NCHUNK) {
      lstore(nxt & 1);                    // other buffer: no conflict w/ readers
      __syncthreads();                    // single barrier per chunk
    }
  }

  // C layout: VGPR j, lanes 0-15 -> M=j, N=lane; lanes 16-31 -> M=j+8.
  const int mbase = mt * BM + wm * 64 + ((lane >> 4) << 3);
  const int qbase = qt * BN + wn * 32 + (lane & 15);
#pragma unroll
  for (int ms = 0; ms < 4; ++ms)
#pragma unroll
    for (int ns = 0; ns < 2; ++ns)
#pragma unroll
      for (int j = 0; j < 8; ++j) {
        int b = mbase + ms * 16 + j;
        int q = qbase + ns * 16;
        int l = q & 3;
        int r = (kblk << 8) + (q >> 2);
        float c = acc[ms][ns][j];
        bf16_t h  = (bf16_t)c;
        bf16_t lo = (bf16_t)(c - (float)h);
        unsigned int pk =
            ((unsigned int)__builtin_bit_cast(unsigned short, h) << 16) |
            (unsigned int)__builtin_bit_cast(unsigned short, lo);
        t[(size_t)b * N_TOT + l * BLK_K + r] = pk;
      }
}

// ---------------------------------------------------------------------------
// Stage 2: out[b, s*4+l] = sum_r t[b,l,r] * w2[l,s,r] + bias[s*4+l]
// ---------------------------------------------------------------------------
__global__ __launch_bounds__(256)
void monarch_stage2_bf16x3(const unsigned int* __restrict__ t,
                           const float* __restrict__ w2,
                           const float* __restrict__ bias,
                           float* __restrict__ out)
{
  __shared__ bf16_t sAhi[2][BM * AS], sAlo[2][BM * AS];
  __shared__ bf16_t sBhi[2][BN * AS], sBlo[2][BN * AS];

  const int tid  = threadIdx.x;
  const int lane = tid & 31;
  const int wave = tid >> 5;
  const int wm   = wave >> 2;
  const int wn   = wave & 3;

  const int st = blockIdx.x;   // N (s) tile
  const int mt = blockIdx.y;   // M tile
  const int l  = blockIdx.z;   // monarch block (0..3)

  const unsigned int* tg = t  + (size_t)mt * BM * N_TOT + (size_t)l * BLK_K;
  const float*        wg = w2 + (size_t)l * BLK_K * BLK_K + (size_t)st * BN * BLK_K;

  const int ldRow = tid >> 3;
  const int ldCol = (tid & 7) << 2;

  const v8f vzero = {0.f, 0.f, 0.f, 0.f, 0.f, 0.f, 0.f, 0.f};
  v8f acc[4][2];
#pragma unroll
  for (int i = 0; i < 4; ++i)
#pragma unroll
    for (int j = 0; j < 2; ++j) acc[i][j] = vzero;

  v4u aR[4];
  v4f bR[4];

  auto gload = [&](int kb) {
#pragma unroll
    for (int it = 0; it < 4; ++it) {
      int row = ldRow + it * 32;
      aR[it] = *(const v4u*)(tg + (size_t)row * N_TOT + kb + ldCol);
      bR[it] = *(const v4f*)(wg + (size_t)row * BLK_K + kb + ldCol);
    }
  };
  auto lstore = [&](int buf) {
#pragma unroll
    for (int it = 0; it < 4; ++it) {
      int idx = (ldRow + it * 32) * AS + ldCol;
#pragma unroll
      for (int j = 0; j < 4; ++j) {
        unsigned int u = aR[it][j];
        sAhi[buf][idx + j] = __builtin_bit_cast(bf16_t, (unsigned short)(u >> 16));
        sAlo[buf][idx + j] = __builtin_bit_cast(bf16_t, (unsigned short)(u & 0xFFFFu));
        split_store(sBhi[buf], sBlo[buf], idx + j, bR[it][j]);
      }
    }
  };
  auto compute = [&](int buf) {
    Frag bhi[2], blo[2];
#pragma unroll
    for (int ns = 0; ns < 2; ++ns) {
      bhi[ns] = frag_b(sBhi[buf], wn * 32 + ns * 16, lane);
      blo[ns] = frag_b(sBlo[buf], wn * 32 + ns * 16, lane);
    }
#pragma unroll
    for (int ms = 0; ms < 4; ++ms) {
      Frag ahi = frag_a(sAhi[buf], wm * 64 + ms * 16, lane);
      Frag alo = frag_a(sAlo[buf], wm * 64 + ms * 16, lane);
#pragma unroll
      for (int ns = 0; ns < 2; ++ns) {
        acc[ms][ns] = wmma_bf16(ahi.v, bhi[ns].v, acc[ms][ns]);
        acc[ms][ns] = wmma_bf16(ahi.v, blo[ns].v, acc[ms][ns]);
        acc[ms][ns] = wmma_bf16(alo.v, bhi[ns].v, acc[ms][ns]);
      }
    }
  };

  gload(0);
  lstore(0);
  __syncthreads();
  for (int c = 0; c < NCHUNK; ++c) {
    const int nxt = c + 1;
    if (nxt < NCHUNK) gload(nxt * KC);
    compute(c & 1);
    if (nxt < NCHUNK) {
      lstore(nxt & 1);
      __syncthreads();
    }
  }

  const int mbase = mt * BM + wm * 64 + ((lane >> 4) << 3);
  const int sbase = st * BN + wn * 32 + (lane & 15);
#pragma unroll
  for (int ms = 0; ms < 4; ++ms)
#pragma unroll
    for (int ns = 0; ns < 2; ++ns)
#pragma unroll
      for (int j = 0; j < 8; ++j) {
        int b = mbase + ms * 16 + j;
        int s = sbase + ns * 16;
        int col = s * NBLK + l;
        out[(size_t)b * N_TOT + col] = acc[ms][ns][j] + bias[col];
      }
}

// ---------------------------------------------------------------------------
extern "C" void kernel_launch(void* const* d_in, const int* in_sizes, int n_in,
                              void* d_out, int out_size, void* d_ws, size_t ws_size,
                              hipStream_t stream) {
  (void)n_in; (void)out_size; (void)ws_size;
  const float* x    = (const float*)d_in[0];
  const float* w1   = (const float*)d_in[1];
  const float* w2   = (const float*)d_in[2];
  const float* bias = (const float*)d_in[3];
  float*        out = (float*)d_out;
  unsigned int* t   = (unsigned int*)d_ws;  // 16384*4096*4B = 256 MB intermediate

  const int Brows = in_sizes[0] / N_TOT;    // 16384
  dim3 block(256, 1, 1);
  dim3 grid(BLK_K / BN, (unsigned)(Brows / BM), NBLK);

  monarch_stage1_bf16x3<<<grid, block, 0, stream>>>(x, w1, t);
  monarch_stage2_bf16x3<<<grid, block, 0, stream>>>(t, w2, bias, out);
}
